// nGPTAttention_68229850464864
// MI455X (gfx1250) — matrix operand order repaired and checked by reading, not compiled, not run on hardware
//
#include <hip/hip_runtime.h>
#include <hip/hip_bf16.h>

typedef __bf16 bf16;
typedef __attribute__((ext_vector_type(16))) __bf16 v16bf;
typedef __attribute__((ext_vector_type(8)))  __bf16 v8bf;
typedef __attribute__((ext_vector_type(2)))  __bf16 v2bf;
typedef __attribute__((ext_vector_type(8)))  float  v8f;
typedef __attribute__((ext_vector_type(4)))  int    v4i;

union BF16x16 { v16bf v; v8bf h[2]; };

#define WMMA_BF16(a, b, c) \
  __builtin_amdgcn_wmma_f32_16x16x32_bf16(false, (a), false, (b), (short)0, (c), false, false)

// ---------------------------------------------------------------------------
// CDNA5 async global->LDS staging (ASYNCcnt path) with safe fallback.
// Probe-confirmed signature: (v4i addrspace(1)*, v4i addrspace(3)*, imm, imm)
// ---------------------------------------------------------------------------
#if defined(__has_builtin)
#if __has_builtin(__builtin_amdgcn_global_load_async_to_lds_b128) && \
    __has_builtin(__builtin_amdgcn_s_wait_asynccnt)
#define HAS_ASYNC_LDS 1
#endif
#endif

typedef __attribute__((address_space(1))) v4i* gptr_v4i;
typedef __attribute__((address_space(3))) v4i* lptr_v4i;

__device__ __forceinline__ void copy16_g2l(const bf16* g, bf16* l) {
#ifdef HAS_ASYNC_LDS
  __builtin_amdgcn_global_load_async_to_lds_b128(
      (gptr_v4i)(uintptr_t)g,
      (lptr_v4i)(unsigned int)(uintptr_t)l, 0, 0);
#else
  *(v8bf*)l = *(const v8bf*)g;
#endif
}

template <int N>
__device__ __forceinline__ void wait_stage() {
#ifdef HAS_ASYNC_LDS
  __builtin_amdgcn_s_wait_asynccnt(N);
#endif
}

// ---------------------------------------------------------------------------
// fp32 -> bf16 elementwise convert
// ---------------------------------------------------------------------------
__global__ void cvt_f32_bf16(const float* __restrict__ in, bf16* __restrict__ out, int n) {
  int i = blockIdx.x * blockDim.x + threadIdx.x;
  if (i < n) out[i] = (bf16)in[i];
}

// ---------------------------------------------------------------------------
// NT GEMM: C[m,n] = sum_k A[m,k] * B[n,k]   (A: MxK row-major, B: NxK row-major)
// Block: 8 waves, one nt (64 cols) shared; wave computes a 16x64 tile.
// B panel (64x32 bf16, 4KB) staged in double-buffered LDS, shared by all waves.
// ---------------------------------------------------------------------------
__global__ void gemm_nt_bf16(const bf16* __restrict__ A, const bf16* __restrict__ Bm,
                             float* __restrict__ Cm, int M, int N, int K) {
  __shared__ __align__(16) bf16 Bs[2][64 * 32];

  const int tid  = threadIdx.x;
  const int lane = tid & 31;
  const int wid  = tid >> 5;
  const int mtGroups = M >> 7;                    // (M/16)/8
  const int nt = blockIdx.x / mtGroups;
  const int mt = (blockIdx.x % mtGroups) * 8 + wid;
  const int m0 = mt << 4;
  const int n0 = nt << 6;
  const int lm   = lane & 15;
  const int half = lane >> 4;

  // cooperative staging map: thread -> (panel row 0..63, 8-elem chunk 0..3)
  const int srow   = tid >> 2;
  const int schunk = (tid & 3) * 8;
  const bf16* bsrc = Bm + (size_t)(n0 + srow) * K + schunk;
  bf16* bdst0 = &Bs[0][srow * 32 + schunk];
  bf16* bdst1 = &Bs[1][srow * 32 + schunk];

  v8f acc[4];
#pragma unroll
  for (int f = 0; f < 4; ++f) acc[f] = (v8f){0.f,0.f,0.f,0.f,0.f,0.f,0.f,0.f};

  const bf16* arow = A + (size_t)(m0 + lm) * K;
  const int nsteps = K >> 5;

  copy16_g2l(bsrc, bdst0);                        // stage step 0

  for (int i = 0; i < nsteps; ++i) {
    const int k0  = i << 5;
    const int cur = i & 1;
    if (i + 1 < nsteps) {
      copy16_g2l(bsrc + ((size_t)(i + 1) << 5), cur ? bdst0 : bdst1);
      wait_stage<1>();                            // step i's panel is resident
    } else {
      wait_stage<0>();
    }

    // wave-private A fragment from global (regular vmem, overlaps staging)
    BF16x16 a;
    a.h[0] = *(const v8bf*)(arow + k0 + 8 * half);
    a.h[1] = *(const v8bf*)(arow + k0 + 16 + 8 * half);

    __syncthreads();
#pragma unroll
    for (int f = 0; f < 4; ++f) {
      const bf16* bp = &Bs[cur][(f * 16 + lm) * 32 + 8 * half];
      BF16x16 b;
      b.h[0] = *(const v8bf*)bp;
      b.h[1] = *(const v8bf*)(bp + 16);
      acc[f] = WMMA_BF16(a.v, b.v, acc[f]);
    }
    __syncthreads();
  }

#pragma unroll
  for (int f = 0; f < 4; ++f)
#pragma unroll
    for (int r = 0; r < 8; ++r)
      Cm[(size_t)(m0 + r + 8 * half) * N + n0 + f * 16 + lm] = acc[f][r];
}

// ---------------------------------------------------------------------------
// RoPE (interleaved pairs) + L2 normalize + s_qk * sqrt(C) scale, fp32 -> bf16.
// One wave per (b,h,t) row; lane owns pair (2*lane, 2*lane+1).
// src: [B*T, C] fp32;  dst: [B*H, T, D] bf16
// ---------------------------------------------------------------------------
__global__ void rope_norm_scale(const float* __restrict__ src, const float* __restrict__ s_qk,
                                bf16* __restrict__ dst, int Hn, int Tn, int Dn, int Cn) {
  const int lane = threadIdx.x & 31;
  const int wid  = blockIdx.x * (blockDim.x >> 5) + (threadIdx.x >> 5);
  const int t = wid % Tn;
  const int h = (wid / Tn) % Hn;
  const int b = wid / (Tn * Hn);
  const int d0 = 2 * lane;

  const float2 xv = *(const float2*)(src + (size_t)(b * Tn + t) * Cn + h * Dn + d0);
  // inv_freq = theta^{-d0/D}, ln(10000) = 9.2103403719761836
  const float ang = (float)t * __expf(-9.2103403719761836f * ((float)d0 / (float)Dn));
  float s, c;
  __sincosf(ang, &s, &c);
  const float y0 = xv.x * c - xv.y * s;
  const float y1 = xv.y * c + xv.x * s;

  float ss = y0 * y0 + y1 * y1;
#pragma unroll
  for (int m = 16; m >= 1; m >>= 1) ss += __shfl_xor(ss, m, 32);
  const float inv = 1.0f / fmaxf(sqrtf(ss), 1e-12f);

  const float sc = 32.0f; // s_qk_init / s_qk_scale = sqrt(C)
  const float g0 = s_qk[h * Dn + d0] * sc * inv;
  const float g1 = s_qk[h * Dn + d0 + 1] * sc * inv;

  v2bf o;
  o[0] = (bf16)(y0 * g0);
  o[1] = (bf16)(y1 * g1);
  *(v2bf*)(dst + ((size_t)(b * Hn + h) * Tn + t) * Dn + d0) = o;
}

// ---------------------------------------------------------------------------
// V: fp32 [B*T, C] -> bf16 transposed [B*H, D, T] so P@V B-operand is contiguous in K
// ---------------------------------------------------------------------------
__global__ void v_cvt_transpose(const float* __restrict__ src, bf16* __restrict__ dst,
                                int Hn, int Tn, int Dn, int Cn) {
  int i = blockIdx.x * blockDim.x + threadIdx.x; // i = ((bh*D + d)*T + t)
  int t  = i % Tn;
  int d  = (i / Tn) % Dn;
  int bh = i / (Tn * Dn);
  int b = bh / Hn, h = bh % Hn;
  dst[i] = (bf16)src[(size_t)(b * Tn + t) * Cn + h * Dn + d];
}

// ---------------------------------------------------------------------------
// Flash attention. Block: 8 waves, one (b,h) shared; wave owns a 16-row q tile.
// K chunk (32x64) and V chunk (64x32) double-buffered in LDS, shared by waves.
// Q,K: [BH, T, 64] bf16 ; Vt: [BH, 64, T] bf16 ; Y: [B*T, C] bf16
// ---------------------------------------------------------------------------
__global__ void attn_flash(const bf16* __restrict__ Q, const bf16* __restrict__ Kt,
                           const bf16* __restrict__ Vt, bf16* __restrict__ Y,
                           int Hn, int Tn) {
  const int Dn = 64;
  __shared__ __align__(16) bf16 Ks[2][32 * 64];
  __shared__ __align__(16) bf16 Vs[2][64 * 32];
  __shared__ __align__(16) bf16 Plds[8][16 * 32];

  const int tid  = threadIdx.x;
  const int lane = tid & 31;
  const int w    = tid >> 5;
  const int ttGroups = Tn >> 7;                   // (T/16)/8
  const int bh = blockIdx.x / ttGroups;
  const int tt = (blockIdx.x % ttGroups) * 8 + w;
  const int lm   = lane & 15;
  const int half = lane >> 4;

  // cooperative staging maps
  const int krow = tid >> 3, kchunk = (tid & 7) * 8;      // K: 32 rows x 64 cols
  const int vrow = tid >> 2, vchunk = (tid & 3) * 8;      // V: 64 rows x 32 cols
  const bf16* ksrc = Kt + ((size_t)bh * Tn + krow) * Dn + kchunk;
  const bf16* vsrc = Vt + ((size_t)bh * Dn + vrow) * Tn + vchunk;
  bf16* kdst0 = &Ks[0][krow * 64 + kchunk];
  bf16* kdst1 = &Ks[1][krow * 64 + kchunk];
  bf16* vdst0 = &Vs[0][vrow * 32 + vchunk];
  bf16* vdst1 = &Vs[1][vrow * 32 + vchunk];

  // q tile -> two 16x32 A fragments
  const size_t qoff = ((size_t)bh * Tn + tt * 16 + lm) * Dn;
  BF16x16 aq[2];
#pragma unroll
  for (int kk = 0; kk < 2; ++kk) {
    aq[kk].h[0] = *(const v8bf*)(Q + qoff + kk * 32 + 8 * half);
    aq[kk].h[1] = *(const v8bf*)(Q + qoff + kk * 32 + 16 + 8 * half);
  }

  v8f acc[4];
#pragma unroll
  for (int f = 0; f < 4; ++f) acc[f] = (v8f){0.f,0.f,0.f,0.f,0.f,0.f,0.f,0.f};
  float mrow[8], lrow[8];
#pragma unroll
  for (int r = 0; r < 8; ++r) { mrow[r] = -1e30f; lrow[r] = 0.f; }

  const int nchunks = Tn >> 5;                    // 32 keys / chunk

  copy16_g2l(ksrc, kdst0);                        // stage chunk 0
  copy16_g2l(vsrc, vdst0);

  for (int i = 0; i < nchunks; ++i) {
    const int cur = i & 1;
    if (i + 1 < nchunks) {
      copy16_g2l(ksrc + (size_t)(i + 1) * 32 * Dn, cur ? kdst0 : kdst1);
      copy16_g2l(vsrc + (size_t)(i + 1) * 32,      cur ? vdst0 : vdst1);
      wait_stage<2>();
    } else {
      wait_stage<0>();
    }
    __syncthreads();

    // S = q @ k^T : two 16x16 tiles, K=64 -> 2 WMMAs each, B frags from LDS
    v8f S[2];
    S[0] = (v8f){0.f,0.f,0.f,0.f,0.f,0.f,0.f,0.f};
    S[1] = (v8f){0.f,0.f,0.f,0.f,0.f,0.f,0.f,0.f};
#pragma unroll
    for (int n2 = 0; n2 < 2; ++n2) {
#pragma unroll
      for (int kk = 0; kk < 2; ++kk) {
        const bf16* kp = &Ks[cur][(n2 * 16 + lm) * 64 + kk * 32 + 8 * half];
        BF16x16 bk;
        bk.h[0] = *(const v8bf*)kp;
        bk.h[1] = *(const v8bf*)(kp + 16);
        S[n2] = WMMA_BF16(aq[kk].v, bk.v, S[n2]);
      }
    }

    // pull V fragments early so LDS loads overlap softmax VALU/trans work
    BF16x16 bv[4];
#pragma unroll
    for (int f = 0; f < 4; ++f) {
      const bf16* vp = &Vs[cur][(f * 16 + lm) * 32 + 8 * half];
      bv[f].h[0] = *(const v8bf*)vp;
      bv[f].h[1] = *(const v8bf*)(vp + 16);
    }

    // online softmax (row state: reg r <-> row r + 8*half)
    float p0[8], p1[8], cm[8];
#pragma unroll
    for (int r = 0; r < 8; ++r) {
      p0[r] = S[0][r] * 8.0f; // * sqrt(D)
      p1[r] = S[1][r] * 8.0f;
      float t2 = fmaxf(p0[r], p1[r]);
      t2 = fmaxf(t2, __shfl_xor(t2, 1, 32));
      t2 = fmaxf(t2, __shfl_xor(t2, 2, 32));
      t2 = fmaxf(t2, __shfl_xor(t2, 4, 32));
      t2 = fmaxf(t2, __shfl_xor(t2, 8, 32));
      cm[r] = t2;
    }
#pragma unroll
    for (int r = 0; r < 8; ++r) {
      const float nm = fmaxf(mrow[r], cm[r]);
      const float al = __expf(mrow[r] - nm);
      mrow[r] = nm;
      p0[r] = __expf(p0[r] - nm);
      p1[r] = __expf(p1[r] - nm);
      float s2 = p0[r] + p1[r];
      s2 += __shfl_xor(s2, 1, 32);
      s2 += __shfl_xor(s2, 2, 32);
      s2 += __shfl_xor(s2, 4, 32);
      s2 += __shfl_xor(s2, 8, 32);
      lrow[r] = lrow[r] * al + s2;
#pragma unroll
      for (int f = 0; f < 4; ++f) acc[f][r] *= al;
      // P (C-layout) -> wave-private LDS, row-major 16x32 bf16
      Plds[w][(r + 8 * half) * 32 + lm]      = (bf16)p0[r];
      Plds[w][(r + 8 * half) * 32 + 16 + lm] = (bf16)p1[r];
    }

    // reload P as A-fragment (16x32), wave-private so DScnt ordering suffices
    BF16x16 pa;
    pa.h[0] = *(const v8bf*)&Plds[w][lm * 32 + 8 * half];
    pa.h[1] = *(const v8bf*)&Plds[w][lm * 32 + 16 + 8 * half];

#pragma unroll
    for (int f = 0; f < 4; ++f) acc[f] = WMMA_BF16(pa.v, bv[f].v, acc[f]);

    __syncthreads();
  }

  // epilogue: divide by row sum, scatter to [B*T, C] bf16 for the output GEMM
  const int b = bh / Hn, h = bh % Hn;
  const int Cn = Hn * Dn;
#pragma unroll
  for (int r = 0; r < 8; ++r) {
    const float inv = 1.0f / lrow[r];
    const size_t yo = (size_t)(b * Tn + tt * 16 + r + 8 * half) * Cn + h * Dn;
#pragma unroll
    for (int f = 0; f < 4; ++f)
      Y[yo + f * 16 + lm] = (bf16)(acc[f][r] * inv);
  }
}

// ---------------------------------------------------------------------------
extern "C" void kernel_launch(void* const* d_in, const int* in_sizes, int n_in,
                              void* d_out, int out_size, void* d_ws, size_t ws_size,
                              hipStream_t stream) {
  const int Bn = 2, Tn = 2048, Cn = 1024, Hn = 16, Dn = 64;
  const int M = Bn * Tn;

  const float* x    = (const float*)d_in[0];
  const float* Wq   = (const float*)d_in[1];
  const float* Wk   = (const float*)d_in[2];
  const float* Wv   = (const float*)d_in[3];
  const float* Wo   = (const float*)d_in[4];
  const float* s_qk = (const float*)d_in[5];

  char* ws = (char*)d_ws;
  size_t off = 0;
  auto alloc = [&](size_t bytes) -> void* {
    void* p = ws + off;
    off += (bytes + 255) & ~(size_t)255;
    return p;
  };

  bf16*  xb  = (bf16*)alloc((size_t)M * Cn * 2);
  bf16*  wqb = (bf16*)alloc((size_t)Cn * Cn * 2);
  bf16*  wkb = (bf16*)alloc((size_t)Cn * Cn * 2);
  bf16*  wvb = (bf16*)alloc((size_t)Cn * Cn * 2);
  bf16*  wob = (bf16*)alloc((size_t)Cn * Cn * 2);
  float* qf  = (float*)alloc((size_t)M * Cn * 4);
  float* kf  = (float*)alloc((size_t)M * Cn * 4);
  float* vf  = (float*)alloc((size_t)M * Cn * 4);
  bf16*  qbh = (bf16*)alloc((size_t)M * Cn * 2);
  bf16*  kbh = (bf16*)alloc((size_t)M * Cn * 2);
  bf16*  vt  = (bf16*)alloc((size_t)M * Cn * 2);
  bf16*  yb  = (bf16*)alloc((size_t)M * Cn * 2);

  // 1) converts
  cvt_f32_bf16<<<(M * Cn) / 256, 256, 0, stream>>>(x, xb, M * Cn);
  cvt_f32_bf16<<<(Cn * Cn) / 256, 256, 0, stream>>>(Wq, wqb, Cn * Cn);
  cvt_f32_bf16<<<(Cn * Cn) / 256, 256, 0, stream>>>(Wk, wkb, Cn * Cn);
  cvt_f32_bf16<<<(Cn * Cn) / 256, 256, 0, stream>>>(Wv, wvb, Cn * Cn);
  cvt_f32_bf16<<<(Cn * Cn) / 256, 256, 0, stream>>>(Wo, wob, Cn * Cn);

  // 2) QKV projections (NT GEMM, LDS-staged B panel, 8 waves / block)
  const int gemmBlocks = (Cn / 64) * (M / 128);
  gemm_nt_bf16<<<gemmBlocks, 256, 0, stream>>>(xb, wqb, qf, M, Cn, Cn);
  gemm_nt_bf16<<<gemmBlocks, 256, 0, stream>>>(xb, wkb, kf, M, Cn, Cn);
  gemm_nt_bf16<<<gemmBlocks, 256, 0, stream>>>(xb, wvb, vf, M, Cn, Cn);

  // 3) RoPE + normalize + scale for q, k
  const int ropeBlocks = (Bn * Hn * Tn) / 8;
  rope_norm_scale<<<ropeBlocks, 256, 0, stream>>>(qf, s_qk, qbh, Hn, Tn, Dn, Cn);
  rope_norm_scale<<<ropeBlocks, 256, 0, stream>>>(kf, s_qk, kbh, Hn, Tn, Dn, Cn);

  // 4) V -> transposed bf16 [BH, D, T]
  v_cvt_transpose<<<(M * Cn) / 256, 256, 0, stream>>>(vf, vt, Hn, Tn, Dn, Cn);

  // 5) attention (K/V chunks LDS-shared across the block's 8 waves)
  const int attnBlocks = (Bn * Hn) * (Tn / 128);
  attn_flash<<<attnBlocks, 256, 0, stream>>>(qbh, kbh, vt, yb, Hn, Tn);

  // 6) output projection -> fp32 d_out
  gemm_nt_bf16<<<gemmBlocks, 256, 0, stream>>>(yb, wob, (float*)d_out, M, Cn, Cn);
}
